// SNNNetwork_14370960572561
// MI455X (gfx1250) — matrix operand order repaired and checked by reading, not compiled, not run on hardware
//
#include <hip/hip_runtime.h>

// ---------------------------------------------------------------------------
// SNN 4-layer scan for MI455X (gfx1250, wave32, WMMA).
//  - 16 workgroups (B/16), 256 threads (8 waves). Each WG owns 16 batch rows
//    and runs the full T=512 scan; spk exchanged via LDS + __syncthreads.
//  - Recurrent GEMMs via v_wmma_f32_16x16x32_bf16 with hi/lo bf16 weight
//    split (spk is exactly representable in bf16) -> ~fp32 accuracy.
//  - Weights pre-packed into WMMA B-fragment order in d_ws (L2-resident).
//  - A laundered ZERO OFFSET (not the pointer) defeats LICM hoisting of the
//    loop-invariant weight loads while preserving pointer provenance, so the
//    backend emits global_load_b128 (LOADcnt only), not flat_load (which
//    couples LOADcnt+DScnt and forced full-drain waits before every WMMA).
//  - All result streams stored non-temporally (537MB write-once).
// ---------------------------------------------------------------------------

typedef __attribute__((ext_vector_type(16))) __bf16        v16bf;
typedef __attribute__((ext_vector_type(8)))  float         v8f;
typedef __attribute__((ext_vector_type(4)))  unsigned int  u32x4;
typedef __attribute__((ext_vector_type(4)))  float         f32x4;

#define BETA 0.8f
#define THRV 1.0f
#define Bb 256
#define Tt 512
#define Ff 8
#define Hh 256
#define Ll 4
#define Oo 8
#define SPK_PITCH 264   // 256 + 8 ushorts: 528B row stride breaks LDS bank-stride conflicts

__device__ __forceinline__ unsigned short f2bf(float f) {
  unsigned int u = __float_as_uint(f);
  u += 0x7FFFu + ((u >> 16) & 1u);          // round-to-nearest-even
  return (unsigned short)(u >> 16);
}
__device__ __forceinline__ float bf2f(unsigned short h) {
  return __uint_as_float(((unsigned int)h) << 16);
}

// ---------------------------------------------------------------------------
// Prep: pack Wh (3,H,H) and zero-padded Wout (H,16) into WMMA B-fragment
// order, bf16 hi/lo split.
// frag = ((layer*16 + ntile)*8 + ktile)*2 + part; each frag = 32 lanes x 16 bf16.
// Lane l, elem j of fragment holds W[k][n] with n = ntile*16 + (l&15),
// k = ktile*32 + (l>>4)*16 + j  (contiguous k order, same as the A-operand LDS loads).
// ---------------------------------------------------------------------------
__global__ void snn_pack_weights(const float* __restrict__ Wh,
                                 const float* __restrict__ Wout,
                                 unsigned short* __restrict__ packed) {
  int e = blockIdx.x * blockDim.x + threadIdx.x;
  if (e >= 1024 * 512) return;
  int frag = e >> 9;
  int r    = e & 511;
  int lane = r >> 4, j = r & 15;
  int part  = frag & 1;
  int kt    = (frag >> 1) & 7;
  int nt    = (frag >> 4) & 15;
  int layer = frag >> 8;
  int n = nt * 16 + (lane & 15);
  int k = kt * 32 + ((lane >> 4) << 4) + j;
  float w;
  if (layer < 3) w = Wh[layer * Hh * Hh + k * Hh + n];
  else           w = (nt == 0 && n < Oo) ? Wout[k * Oo + n] : 0.0f;
  unsigned short hi  = f2bf(w);
  unsigned short val = part ? f2bf(w - bf2f(hi)) : hi;
  packed[frag * 512 + lane * 16 + j] = val;
}

__global__ void snn_copy(const float* __restrict__ src, float* __restrict__ dst, int n) {
  int i = blockIdx.x * blockDim.x + threadIdx.x;
  if (i < n) __builtin_nontemporal_store(src[i], &dst[i]);
}

// ---------------------------------------------------------------------------
// Main scan kernel. grid = 16, block = 256 (8 waves).
// Wave w owns output N-tiles {2w, 2w+1} (32 hidden units) for all 4 layers.
// ---------------------------------------------------------------------------
__global__ void __launch_bounds__(256)
snn_scan(const float* __restrict__ x,
         const float* __restrict__ hs0,
         const float* __restrict__ rs0,
         const float* __restrict__ W1,
         const float* __restrict__ b1,
         const float* __restrict__ bh,
         const float* __restrict__ bout,
         const unsigned short* __restrict__ packedW,
         float* __restrict__ outRo,       // (B,T,O)
         float* __restrict__ outHidden,   // (B,T,L,H)
         float* __restrict__ outRo2)      // (B,T,O)
{
  __shared__ __align__(16) unsigned short spk[Ll][16][SPK_PITCH];  // ~33 KB

  const int tid  = threadIdx.x;
  const int wv   = tid >> 5;
  const int lane = tid & 31;
  const int hl   = lane >> 4;      // lane half: selects A K-range / C row set
  const int ln   = lane & 15;      // column within N-tile (C/D N = ln)
  const int b0   = blockIdx.x * 16;

  const int nt0g = wv * 2, nt1g = wv * 2 + 1;
  const int n0 = nt0g * 16 + ln, n1 = nt1g * 16 + ln;

  // ---- loop-invariant params in registers (exact fp32) ----
  float w1r[2][Ff];
  #pragma unroll
  for (int f = 0; f < Ff; ++f) {
    w1r[0][f] = W1[f * Hh + n0];
    w1r[1][f] = W1[f * Hh + n1];
  }
  const float b1r0 = b1[n0], b1r1 = b1[n1];
  float bhr[3][2];
  #pragma unroll
  for (int i = 0; i < 3; ++i) {
    bhr[i][0] = bh[i * Hh + n0];
    bhr[i][1] = bh[i * Hh + n1];
  }
  const float boutr = (wv == 0 && ln < Oo) ? bout[ln] : 0.0f;

  // ---- state init from inputs (t=0 slices) ----
  float mem[Ll][2][8];
  #pragma unroll
  for (int i = 0; i < Ll; ++i)
    #pragma unroll
    for (int v = 0; v < 8; ++v) {
      const size_t base = (size_t)(b0 + hl * 8 + v) * Tt * (Ll * Hh) + (size_t)i * Hh;
      mem[i][0][v] = hs0[base + n0];
      mem[i][1][v] = hs0[base + n1];
    }
  float ro[8];
  #pragma unroll
  for (int v = 0; v < 8; ++v)
    ro[v] = (wv == 0 && ln < Oo) ? rs0[(size_t)(b0 + hl * 8 + v) * Tt * Oo + ln] : 0.0f;

  // ---- sequential scan ----
  for (int t = 0; t < Tt; ++t) {
    // layer-0 input current: x @ W1 + b1, exact fp32 (K=8, VALU)
    float cur[2][8];
    #pragma unroll
    for (int v = 0; v < 8; ++v) {
      const f32x4* xr = (const f32x4*)(x + ((size_t)(b0 + hl * 8 + v) * Tt + t) * Ff);
      const f32x4 xa = xr[0], xb = xr[1];
      float c0 = b1r0, c1 = b1r1;
      c0 += xa.x * w1r[0][0] + xa.y * w1r[0][1] + xa.z * w1r[0][2] + xa.w * w1r[0][3]
          + xb.x * w1r[0][4] + xb.y * w1r[0][5] + xb.z * w1r[0][6] + xb.w * w1r[0][7];
      c1 += xa.x * w1r[1][0] + xa.y * w1r[1][1] + xa.z * w1r[1][2] + xa.w * w1r[1][3]
          + xb.x * w1r[1][4] + xb.y * w1r[1][5] + xb.z * w1r[1][6] + xb.w * w1r[1][7];
      cur[0][v] = c0;
      cur[1][v] = c1;
    }

    #pragma unroll
    for (int i = 0; i < Ll; ++i) {
      // membrane update + spike (C/D layout) + hidden-state store (NT)
      #pragma unroll
      for (int nt = 0; nt < 2; ++nt) {
        const int n = nt ? n1 : n0;
        #pragma unroll
        for (int v = 0; v < 8; ++v) {
          const float m0 = mem[i][nt][v];
          const float mn = BETA * m0 + cur[nt][v] - (m0 > THRV ? THRV : 0.0f);
          mem[i][nt][v] = mn;
          spk[i][hl * 8 + v][n] = (mn > THRV) ? (unsigned short)0x3F80u
                                              : (unsigned short)0u;
          __builtin_nontemporal_store(
              mn, &outHidden[(((size_t)(b0 + hl * 8 + v) * Tt + t) * Ll + i) * Hh + n]);
        }
      }
      __syncthreads();

      if (i < 3) {
        // cur_{i+1} = spk_i @ Wh[i] + bh[i]  via bf16 hi/lo WMMA.
        // Laundered zero offset: addresses become loop-variant (no LICM/spill)
        // but provenance stays with packedW -> global_load_b128 selection.
        size_t zoff = 0;
        asm volatile("" : "+s"(zoff));
        const unsigned short* pw = packedW + zoff;
        v8f acc0 = {0, 0, 0, 0, 0, 0, 0, 0};
        v8f acc1 = {0, 0, 0, 0, 0, 0, 0, 0};
        const unsigned short* srow = &spk[i][ln][0];
        #pragma unroll
        for (int kt = 0; kt < 8; ++kt) {
          union { u32x4 q[2]; v16bf v; } a;
          a.q[0] = *(const u32x4*)(srow + kt * 32 + hl * 16);        // K run 0
          a.q[1] = *(const u32x4*)(srow + kt * 32 + hl * 16 + 8);   // K run 1
          const size_t f0 = (size_t)(((i * 16 + nt0g) * 8 + kt) * 2) * 512 + lane * 16;
          const size_t f1 = (size_t)(((i * 16 + nt1g) * 8 + kt) * 2) * 512 + lane * 16;
          const v16bf bh0 = *(const v16bf*)(pw + f0);
          const v16bf bl0 = *(const v16bf*)(pw + f0 + 512);
          const v16bf bh1 = *(const v16bf*)(pw + f1);
          const v16bf bl1 = *(const v16bf*)(pw + f1 + 512);
          acc0 = __builtin_amdgcn_wmma_f32_16x16x32_bf16(false, a.v, false, bh0, (short)0, acc0, false, false);
          acc0 = __builtin_amdgcn_wmma_f32_16x16x32_bf16(false, a.v, false, bl0, (short)0, acc0, false, false);
          acc1 = __builtin_amdgcn_wmma_f32_16x16x32_bf16(false, a.v, false, bh1, (short)0, acc1, false, false);
          acc1 = __builtin_amdgcn_wmma_f32_16x16x32_bf16(false, a.v, false, bl1, (short)0, acc1, false, false);
        }
        #pragma unroll
        for (int v = 0; v < 8; ++v) {
          cur[0][v] = acc0[v] + bhr[i][0];
          cur[1][v] = acc1[v] + bhr[i][1];
        }
      } else if (wv == 0) {
        // readout GEMM: spk_3 @ Wout (padded to 16 cols) -> ro update + store
        size_t zoff = 0;
        asm volatile("" : "+s"(zoff));
        const unsigned short* pw = packedW + zoff;
        v8f acc = {0, 0, 0, 0, 0, 0, 0, 0};
        const unsigned short* srow = &spk[3][ln][0];
        #pragma unroll
        for (int kt = 0; kt < 8; ++kt) {
          union { u32x4 q[2]; v16bf v; } a;
          a.q[0] = *(const u32x4*)(srow + kt * 32 + hl * 16);
          a.q[1] = *(const u32x4*)(srow + kt * 32 + hl * 16 + 8);
          const size_t f3 = (size_t)(((3 * 16 + 0) * 8 + kt) * 2) * 512 + lane * 16;
          const v16bf bhv = *(const v16bf*)(pw + f3);
          const v16bf blv = *(const v16bf*)(pw + f3 + 512);
          acc = __builtin_amdgcn_wmma_f32_16x16x32_bf16(false, a.v, false, bhv, (short)0, acc, false, false);
          acc = __builtin_amdgcn_wmma_f32_16x16x32_bf16(false, a.v, false, blv, (short)0, acc, false, false);
        }
        #pragma unroll
        for (int v = 0; v < 8; ++v)
          ro[v] = BETA * ro[v] + (acc[v] + boutr);
        if (ln < Oo) {
          #pragma unroll
          for (int v = 0; v < 8; ++v) {
            const size_t oi = ((size_t)(b0 + hl * 8 + v) * Tt + t) * Oo + ln;
            __builtin_nontemporal_store(ro[v], &outRo[oi]);
            __builtin_nontemporal_store(ro[v], &outRo2[oi]);
          }
        }
      }
    }
  }
}

extern "C" void kernel_launch(void* const* d_in, const int* in_sizes, int n_in,
                              void* d_out, int out_size, void* d_ws, size_t ws_size,
                              hipStream_t stream) {
  const float* x    = (const float*)d_in[0];
  const float* hs0  = (const float*)d_in[1];
  // d_in[2] prev_obs input is unused by the math
  const float* rs0  = (const float*)d_in[3];
  const float* W1   = (const float*)d_in[4];
  const float* b1   = (const float*)d_in[5];
  const float* Wh   = (const float*)d_in[6];
  const float* bh   = (const float*)d_in[7];
  const float* Wout = (const float*)d_in[8];
  const float* bout = (const float*)d_in[9];

  float* out       = (float*)d_out;
  float* outRo     = out;                                        // (B,T,O)
  float* outHidden = out + (size_t)Bb * Tt * Oo;                 // (B,T,L,H)
  float* outPrev   = outHidden + (size_t)Bb * Tt * Ll * Hh;      // (B,T,F)
  float* outRo2    = outPrev + (size_t)Bb * Tt * Ff;             // (B,T,O)

  unsigned short* packed = (unsigned short*)d_ws;                // 1 MB needed

  snn_pack_weights<<<(1024 * 512 + 255) / 256, 256, 0, stream>>>(Wh, Wout, packed);
  snn_copy<<<(Bb * Tt * Ff + 255) / 256, 256, 0, stream>>>(x, outPrev, Bb * Tt * Ff);
  snn_scan<<<Bb / 16, 256, 0, stream>>>(x, hs0, rs0, W1, b1, bh, bout, packed,
                                        outRo, outHidden, outRo2);
}
